// GraphSAGE_12618613915872
// MI455X (gfx1250) — matrix operand order repaired and checked by reading, not compiled, not run on hardware
//
#include <hip/hip_runtime.h>
#include <hip/hip_bf16.h>

// GraphSAGE (pool aggregator) forward, 3 layers, N=50000, E=600000, D=128.
// WMMA f16->f32 GEMMs, u32-atomic segment-max (messages are >= 0), two-stage
// deterministic BatchNorm reduction, fused BN+ELU epilogue.

#define NN 50000
#define EE 600000
#define DD 128
#define LDSTR 136            // padded LDS row stride (halves) to avoid bank conflicts
#define BN_EPS 1e-5f
#define STATS_BLOCKS 400

typedef __attribute__((ext_vector_type(16))) _Float16 v16h;
typedef __attribute__((ext_vector_type(8)))  _Float16 v8h;
typedef __attribute__((ext_vector_type(8)))  float    v8f;

// ---- A fragment: 16x32 f16, row M = lane&15, per ISA 16-bit A layout ----
// lanes 0-15:  v0..3 = K 0..7,  v4..7 = K 16..23   (relative to kb)
// lanes 16-31: v0..3 = K 8..15, v4..7 = K 24..31
__device__ __forceinline__ v16h load_a_frag(const float* __restrict__ xrow,
                                            int kb, int half_) {
  const float4* p0 = (const float4*)(xrow + kb + 8 * half_);
  const float4* p1 = (const float4*)(xrow + kb + 16 + 8 * half_);
  float4 g0 = p0[0], g1 = p0[1], g2 = p1[0], g3 = p1[1];
  v16h a;
  a[0]  = (_Float16)g0.x; a[1]  = (_Float16)g0.y; a[2]  = (_Float16)g0.z; a[3]  = (_Float16)g0.w;
  a[4]  = (_Float16)g1.x; a[5]  = (_Float16)g1.y; a[6]  = (_Float16)g1.z; a[7]  = (_Float16)g1.w;
  a[8]  = (_Float16)g2.x; a[9]  = (_Float16)g2.y; a[10] = (_Float16)g2.z; a[11] = (_Float16)g2.w;
  a[12] = (_Float16)g3.x; a[13] = (_Float16)g3.y; a[14] = (_Float16)g3.z; a[15] = (_Float16)g3.w;
  return a;
}

// ---- B fragment: 32x16 f16 from LDS Wt[col][k] (transposed weights) ----
// lanes 0-15 hold K 0..7,16..23 of column n, lanes 16-31 hold K 8..15,24..31.
__device__ __forceinline__ v16h load_b_frag(const _Float16* Wt, int col,
                                            int kb, int half_) {
  const _Float16* p = Wt + col * LDSTR + kb + 8 * half_;
  v8h b0 = *(const v8h*)p, b1 = *(const v8h*)(p + 16);
  v16h b;
#pragma unroll
  for (int i = 0; i < 8; ++i) { b[i] = b0[i]; b[8 + i] = b1[i]; }
  return b;
}

// MODE 0: out = relu(x@W + bvec)      (fc_pool)
// MODE 1: out = x@W + bvec            (x@Wself + bias)
// MODE 2: out += x@W                  (+= pooled@Wneigh)
template <int MODE>
__global__ __launch_bounds__(256)
void sage_gemm(const float* __restrict__ x, const float* __restrict__ W,
               const float* __restrict__ bvec, float* __restrict__ out) {
  __shared__ _Float16 Wt[DD * LDSTR];
  const int t = threadIdx.x;
  // Transpose W[k][c] -> Wt[c][k] as f16 (coalesced global reads).
  for (int i = t; i < DD * DD; i += 256) {
    int k = i >> 7, c = i & 127;
    Wt[c * LDSTR + k] = (_Float16)W[i];
  }
  __syncthreads();

  const int wave = t >> 5, lane = t & 31;
  const int n16 = lane & 15, half_ = lane >> 4;
  const int rowbase = blockIdx.x * 128 + wave * 16;
  int rA = rowbase + n16;
  if (rA >= NN) rA = NN - 1;                  // clamp loads; stores are guarded
  const float* xrow = x + (size_t)rA * DD;

  v8f acc[8] = {};
  for (int kb = 0; kb < DD; kb += 32) {
    v16h a = load_a_frag(xrow, kb, half_);
#pragma unroll
    for (int ct = 0; ct < 8; ++ct) {
      v16h b = load_b_frag(Wt, ct * 16 + n16, kb, half_);
      acc[ct] = __builtin_amdgcn_wmma_f32_16x16x32_f16(
          false, a, false, b, (short)0, acc[ct], false, false);
    }
  }

#pragma unroll
  for (int ct = 0; ct < 8; ++ct) {
    const int col = ct * 16 + n16;
    const float bb = (MODE == 2) ? 0.f : bvec[col];
#pragma unroll
    for (int r = 0; r < 8; ++r) {
      const int row = rowbase + half_ * 8 + r;   // C/D layout: VGPR r -> M = 8*half + r
      if (row < NN) {
        size_t idx = (size_t)row * DD + col;
        float v = acc[ct][r];
        if (MODE == 0)      out[idx] = fmaxf(v + bb, 0.f);
        else if (MODE == 1) out[idx] = v + bb;
        else                out[idx] += v;
      }
    }
  }
}

__global__ __launch_bounds__(256)
void zero_f32(float* __restrict__ p, int n) {
  int i = blockIdx.x * 256 + threadIdx.x;
  if (i < n) p[i] = 0.f;
}

// One wave per edge: lane l handles features [4l, 4l+4).  msgs >= 0, so
// float max == u32 max on bit patterns, and init-to-0 implements neginf->0.
__global__ __launch_bounds__(256)
void sage_scatter_max(const float* __restrict__ h, const float* __restrict__ ew,
                      const int* __restrict__ src, const int* __restrict__ dst,
                      float* __restrict__ pooled) {
  int t = blockIdx.x * 256 + threadIdx.x;
  int e = t >> 5;
  if (e >= EE) return;
  int lane = t & 31;
  int s = src[e], d = dst[e];
  float wgt = ew[e];
  float4 v = ((const float4*)(h + (size_t)s * DD))[lane];
  unsigned int* p = (unsigned int*)(pooled + (size_t)d * DD) + lane * 4;
  atomicMax(p + 0, __float_as_uint(v.x * wgt));
  atomicMax(p + 1, __float_as_uint(v.y * wgt));
  atomicMax(p + 2, __float_as_uint(v.z * wgt));
  atomicMax(p + 3, __float_as_uint(v.w * wgt));
}

// Stage 1: per-block partial column sums (coalesced, fixed order).
__global__ __launch_bounds__(128)
void bn_stats_partial(const float* __restrict__ rst,
                      float* __restrict__ psum, float* __restrict__ psq) {
  int f = threadIdx.x;
  float s = 0.f, q = 0.f;
  for (int r = blockIdx.x; r < NN; r += STATS_BLOCKS) {
    float v = rst[(size_t)r * DD + f];
    s += v; q += v * v;
  }
  psum[blockIdx.x * DD + f] = s;
  psq[blockIdx.x * DD + f]  = q;
}

// Stage 2: deterministic final reduce -> mean, rstd.
__global__ __launch_bounds__(128)
void bn_stats_final(const float* __restrict__ psum, const float* __restrict__ psq,
                    float* __restrict__ mean, float* __restrict__ rstd) {
  int f = threadIdx.x;
  float s = 0.f, q = 0.f;
  for (int b = 0; b < STATS_BLOCKS; ++b) { s += psum[b * DD + f]; q += psq[b * DD + f]; }
  float m = s / (float)NN;
  float var = q / (float)NN - m * m;     // biased variance, matches jnp.var
  mean[f] = m;
  rstd[f] = rsqrtf(var + BN_EPS);
}

__global__ __launch_bounds__(256)
void bn_elu(const float* __restrict__ rst, const float* __restrict__ mean,
            const float* __restrict__ rstd, const float* __restrict__ gamma,
            const float* __restrict__ beta, float* __restrict__ out) {
  size_t i = (size_t)blockIdx.x * 256 + threadIdx.x;
  if (i >= (size_t)NN * DD) return;
  int f = (int)(i & (DD - 1));
  float z = gamma[f] * (rst[i] - mean[f]) * rstd[f] + beta[f];
  out[i] = (z > 0.f) ? z : expm1f(z);
}

extern "C" void kernel_launch(void* const* d_in, const int* in_sizes, int n_in,
                              void* d_out, int out_size, void* d_ws, size_t ws_size,
                              hipStream_t stream) {
  const float* node  = (const float*)d_in[0];
  const float* ew    = (const float*)d_in[1];
  const float* Wpool = (const float*)d_in[2];
  const float* bpool = (const float*)d_in[3];
  const float* Wneigh= (const float*)d_in[4];
  const float* Wself = (const float*)d_in[5];
  const float* bias  = (const float*)d_in[6];
  const float* gamma = (const float*)d_in[7];
  const float* beta  = (const float*)d_in[8];
  const int*   src   = (const int*)d_in[9];
  const int*   dst   = (const int*)d_in[10];

  const size_t ND = (size_t)NN * DD;
  float* ws    = (float*)d_ws;          // needs ~103 MB of workspace
  float* xbuf0 = ws;
  float* xbuf1 = ws + ND;
  float* hbuf  = ws + 2 * ND;           // h, then reused for rst
  float* pooled= ws + 3 * ND;
  float* psum  = ws + 4 * ND;
  float* psq   = psum + (size_t)STATS_BLOCKS * DD;
  float* meanb = psq  + (size_t)STATS_BLOCKS * DD;
  float* rstdb = meanb + DD;

  const int gemmGrid = (NN + 127) / 128;          // 391
  const int ndGrid   = (int)((ND + 255) / 256);   // 25000
  const int scatGrid = (int)(((size_t)EE * 32 + 255) / 256);

  const float* xin = node;
  for (int l = 0; l < 3; ++l) {
    float* xout = (l == 0) ? xbuf0 : (l == 1) ? xbuf1 : (float*)d_out;
    const size_t wo = (size_t)l * DD * DD;

    // h = relu(x @ Wpool + bpool)
    sage_gemm<0><<<gemmGrid, 256, 0, stream>>>(xin, Wpool + wo, bpool + l * DD, hbuf);
    // pooled = segment_max(h[src]*ew, dst), zero-degree -> 0
    zero_f32<<<ndGrid, 256, 0, stream>>>(pooled, (int)ND);
    sage_scatter_max<<<scatGrid, 256, 0, stream>>>(hbuf, ew, src, dst, pooled);
    // rst = x @ Wself + pooled @ Wneigh + bias   (hbuf reused as rst)
    sage_gemm<1><<<gemmGrid, 256, 0, stream>>>(xin, Wself + wo, bias + l * DD, hbuf);
    sage_gemm<2><<<gemmGrid, 256, 0, stream>>>(pooled, Wneigh + wo, bias + l * DD, hbuf);
    // BatchNorm (training stats) + ELU
    bn_stats_partial<<<STATS_BLOCKS, 128, 0, stream>>>(hbuf, psum, psq);
    bn_stats_final<<<1, 128, 0, stream>>>(psum, psq, meanb, rstdb);
    bn_elu<<<ndGrid, 256, 0, stream>>>(hbuf, meanb, rstdb, gamma + l * DD, beta + l * DD, xout);

    xin = xout;
  }
}
